// LinearAttention_70480413327402
// MI455X (gfx1250) — matrix operand order repaired and checked by reading, not compiled, not run on hardware
//
#include <hip/hip_runtime.h>
#include <math.h>

#define B_ 4
#define F_ 1024
#define S_ 2048
#define I_ 2048
#define E_ 8
#define K_ 5
#define G_ 3
#define T_ (B_ * S_)      // 8192 tokens
#define GF_ (G_ * F_)     // 3072

typedef __attribute__((ext_vector_type(16))) __bf16 v16bf;
typedef __attribute__((ext_vector_type(8)))  __bf16 v8bf;
typedef __attribute__((ext_vector_type(8)))  float  v8f;

// ---------------- conversion / layout kernels ----------------

// inp [B,F,S] f32 -> xbf [T,F] bf16 (token-major)
__global__ void cvt_x_tr_kernel(const float* __restrict__ x, __bf16* __restrict__ dst) {
  size_t d = (size_t)blockIdx.x * 256 + threadIdx.x;   // d < T_*F_
  int f = (int)(d % F_);
  size_t t = d / F_;
  int b = (int)(t / S_), s = (int)(t % S_);
  dst[d] = (__bf16)x[((size_t)b * F_ + f) * S_ + s];
}

// W [E][R][N] f32 -> Wt [E][N][R] bf16 (column-major per expert => contiguous B fragments)
__global__ void cvt_w_tr_kernel(const float* __restrict__ w, __bf16* __restrict__ dst,
                                int R, int N) {
  size_t d = (size_t)blockIdx.x * 256 + threadIdx.x;   // d < E_*R*N
  int r = (int)(d % R);
  size_t en = d / R;
  int n = (int)(en % N);
  int e = (int)(en / N);
  dst[d] = (__bf16)w[((size_t)e * R + r) * N + n];
}

// w1 [O=I][C=I][K] f32 -> w1t [o][k*I+c] bf16 (column-major over reduction dim r=(k,c))
__global__ void cvt_w1_tr_kernel(const float* __restrict__ w1, __bf16* __restrict__ dst) {
  size_t d = (size_t)blockIdx.x * 256 + threadIdx.x;   // d < I_*K_*I_
  size_t rk = d % ((size_t)K_ * I_);
  int o = (int)(d / ((size_t)K_ * I_));
  int k = (int)(rk / I_);
  int c = (int)(rk % I_);
  dst[d] = (__bf16)w1[((size_t)o * I_ + c) * K_ + k];
}

// ---------------- gating / argmax kernels (one wave32 per token) ----------------

__global__ void argmax0_kernel(const float* __restrict__ x, const float* __restrict__ gate,
                               int* __restrict__ idx) {
  int t = blockIdx.x * 8 + (threadIdx.x >> 5);
  int lane = threadIdx.x & 31;
  int b = t / S_, s = t % S_;
  float acc[E_];
#pragma unroll
  for (int e = 0; e < E_; ++e) acc[e] = 0.f;
  for (int f = lane; f < F_; f += 32) {
    float xv = x[((size_t)b * F_ + f) * S_ + s];
#pragma unroll
    for (int e = 0; e < E_; ++e) acc[e] += xv * gate[e * F_ + f];
  }
#pragma unroll
  for (int o2 = 16; o2; o2 >>= 1)
#pragma unroll
    for (int e = 0; e < E_; ++e) acc[e] += __shfl_xor(acc[e], o2, 32);
  if (lane == 0) {
    int best = 0; float bv = acc[0];
#pragma unroll
    for (int e = 1; e < E_; ++e) if (acc[e] > bv) { bv = acc[e]; best = e; }
    idx[t] = best;
  }
}

__global__ void argmax2_kernel(const __bf16* __restrict__ h, const float* __restrict__ gate,
                               int* __restrict__ idx) {
  int t = blockIdx.x * 8 + (threadIdx.x >> 5);
  int lane = threadIdx.x & 31;
  const __bf16* hp = h + (size_t)t * I_;
  float acc[E_];
#pragma unroll
  for (int e = 0; e < E_; ++e) acc[e] = 0.f;
  for (int f = lane; f < I_; f += 32) {
    float xv = (float)hp[f];
#pragma unroll
    for (int e = 0; e < E_; ++e) acc[e] += xv * gate[e * I_ + f];
  }
#pragma unroll
  for (int o2 = 16; o2; o2 >>= 1)
#pragma unroll
    for (int e = 0; e < E_; ++e) acc[e] += __shfl_xor(acc[e], o2, 32);
  if (lane == 0) {
    int best = 0; float bv = acc[0];
#pragma unroll
    for (int e = 1; e < E_; ++e) if (acc[e] > bv) { bv = acc[e]; best = e; }
    idx[t] = best;
  }
}

// ---------------- fragment helpers ----------------

__device__ __forceinline__ v16bf pack_a(v8bf lo, v8bf hi) {
  return __builtin_shufflevector(lo, hi, 0, 1, 2, 3, 4, 5, 6, 7,
                                 8, 9, 10, 11, 12, 13, 14, 15);
}

#define WMMA_BF16(Acc, Av, Bv) \
  __builtin_amdgcn_wmma_f32_16x16x32_bf16(false, (Av), false, (Bv), (short)0, (Acc), false, false)

// ---------------- MoE GEMM (masked expert passes, WMMA bf16, 2x2 register blocking) ----
// A [T, Fin] bf16 token-major; Wt [E][Nout][Fin] bf16 column-major; per-token expert idx.
// Block: 256 thr = 8 waves; block tile = 32 tokens x 256 cols; wave tile = 32x32 (4 WMMA C tiles).
// Both 16x32 A tiles staged per K-step by wave 0 via async global->LDS (ASYNCcnt).
__global__ void moe_gemm_kernel(const __bf16* __restrict__ A, const __bf16* __restrict__ Wt,
                                const int* __restrict__ idx,
                                __bf16* __restrict__ outB, float* __restrict__ outF,
                                int Fin, int Nout) {
  __shared__ __bf16 As[2][16][32];                  // 2KB, tile stride 1KB, row stride 64B
  int t0 = blockIdx.x * 32;
  int wave = threadIdx.x >> 5;
  int lane = threadIdx.x & 31;
  int kh = lane >> 4;                               // K-half select
  int hl = lane & 15;                               // A row / B column within tile
  int n0 = blockIdx.y * 256 + wave * 32;

  int myIdx0 = idx[t0 + hl];                        // experts of this lane's two A rows
  int myIdx1 = idx[t0 + 16 + hl];
  unsigned active = 0;
#pragma unroll
  for (int e = 0; e < E_; ++e)
    if (__ballot(myIdx0 == e) | __ballot(myIdx1 == e)) active |= (1u << e);

  unsigned asBase = (unsigned)(size_t)&As[0][0][0]; // raw LDS offset (low 32b of generic ptr)
  v16bf zv = {};
  v8f z8 = {0.f, 0.f, 0.f, 0.f, 0.f, 0.f, 0.f, 0.f};
  v8f acc[2][2];
#pragma unroll
  for (int tm = 0; tm < 2; ++tm)
#pragma unroll
    for (int tn = 0; tn < 2; ++tn) acc[tm][tn] = z8;

  for (int k0 = 0; k0 < Fin; k0 += 32) {
    if (wave == 0) {                                // async-stage both 16x32 A tiles (2KB)
      const char* g = (const char*)(A + (size_t)(t0 + lane) * Fin + k0);  // 64B per token row
      unsigned l = asBase + lane * 64;              // lane -> (tile=lane>>4, row=lane&15)
      asm volatile("global_load_async_to_lds_b128 %0, %1, off"           :: "v"(l), "v"(g) : "memory");
      asm volatile("global_load_async_to_lds_b128 %0, %1, off offset:16" :: "v"(l), "v"(g) : "memory");
      asm volatile("global_load_async_to_lds_b128 %0, %1, off offset:32" :: "v"(l), "v"(g) : "memory");
      asm volatile("global_load_async_to_lds_b128 %0, %1, off offset:48" :: "v"(l), "v"(g) : "memory");
    }
    asm volatile("s_wait_asynccnt 0" ::: "memory");
    __syncthreads();

    const char* ar0 = (const char*)&As[0][0][0] + hl * 64 + kh * 32;
    v16bf a0 = pack_a(*(const v8bf*)ar0, *(const v8bf*)(ar0 + 16));
    v16bf a1 = pack_a(*(const v8bf*)(ar0 + 1024), *(const v8bf*)(ar0 + 1024 + 16));

    unsigned rem = active;
    while (rem) {                                   // iterate only present experts
      int e = __ffs(rem) - 1;
      rem &= rem - 1;
      v16bf av0 = (myIdx0 == e) ? a0 : zv;          // per-lane row masks, EXEC stays all-1s
      v16bf av1 = (myIdx1 == e) ? a1 : zv;
      const __bf16* bbase = Wt + ((size_t)e * Nout + n0 + hl) * Fin + k0 + kh * 16;
      __builtin_prefetch(bbase + 256, 0, 1);
      v16bf bv0 = *(const v16bf*)bbase;             // contiguous 32B column fragments
      v16bf bv1 = *(const v16bf*)(bbase + (size_t)16 * Fin);
      acc[0][0] = WMMA_BF16(acc[0][0], av0, bv0);
      acc[1][0] = WMMA_BF16(acc[1][0], av1, bv0);
      acc[0][1] = WMMA_BF16(acc[0][1], av0, bv1);
      acc[1][1] = WMMA_BF16(acc[1][1], av1, bv1);
    }
    __syncthreads();
  }

#pragma unroll
  for (int tm = 0; tm < 2; ++tm)
#pragma unroll
    for (int tn = 0; tn < 2; ++tn)
#pragma unroll
      for (int j = 0; j < 8; ++j) {                 // C: VGPR j -> row j + 8*kh
        int m = t0 + tm * 16 + j + kh * 8;
        int n = n0 + tn * 16 + hl;
        float v = acc[tm][tn][j];
        size_t o = (size_t)m * Nout + n;
        if (outB) { v = v > 0.f ? v : 0.f; outB[o] = (__bf16)v; }
        else      { outF[o] = v; }
      }
}

// ---------------- causal conv as implicit GEMM (WMMA bf16, 2x2 register blocking) --------
// H1 [T,I] bf16 token-major; W1t [o][k*I+c] bf16 column-major; H2 = relu(conv) [T,I] bf16.
__global__ void conv_gemm_kernel(const __bf16* __restrict__ H1, const __bf16* __restrict__ W1t,
                                 __bf16* __restrict__ H2) {
  int t0 = blockIdx.x * 32;
  int b = t0 / S_, s0 = t0 % S_;                    // 32-token tiles never cross batch
  int wave = threadIdx.x >> 5;
  int lane = threadIdx.x & 31;
  int kh = lane >> 4, hl = lane & 15;
  int n0 = blockIdx.y * 256 + wave * 32;

  v16bf zv = {};
  v8f z8 = {0.f, 0.f, 0.f, 0.f, 0.f, 0.f, 0.f, 0.f};
  v8f acc[2][2];
#pragma unroll
  for (int tm = 0; tm < 2; ++tm)
#pragma unroll
    for (int tn = 0; tn < 2; ++tn) acc[tm][tn] = z8;

#pragma unroll
  for (int k = 0; k < K_; ++k) {
    int sp0 = s0 + hl + k - (K_ - 1);               // low tile may hit causal zero-pad
    int sp1 = sp0 + 16;                             // always >= 12, never padded
    const __bf16* arow0 = (sp0 >= 0) ? (H1 + ((size_t)b * S_ + sp0) * I_) : nullptr;
    const __bf16* arow1 = H1 + ((size_t)b * S_ + sp1) * I_;
    for (int c0 = 0; c0 < I_; c0 += 32) {
      v16bf av0;
      if (arow0) {
        const __bf16* ap = arow0 + c0 + kh * 16;
        av0 = pack_a(*(const v8bf*)ap, *(const v8bf*)(ap + 8));
      } else {
        av0 = zv;
      }
      const __bf16* ap1 = arow1 + c0 + kh * 16;
      v16bf av1 = pack_a(*(const v8bf*)ap1, *(const v8bf*)(ap1 + 8));

      const __bf16* bbase = W1t + (size_t)(n0 + hl) * (K_ * I_) + (k * I_ + c0) + kh * 16;
      __builtin_prefetch(bbase + 256, 0, 1);
      v16bf bv0 = *(const v16bf*)bbase;
      v16bf bv1 = *(const v16bf*)(bbase + (size_t)16 * (K_ * I_));
      acc[0][0] = WMMA_BF16(acc[0][0], av0, bv0);
      acc[1][0] = WMMA_BF16(acc[1][0], av1, bv0);
      acc[0][1] = WMMA_BF16(acc[0][1], av0, bv1);
      acc[1][1] = WMMA_BF16(acc[1][1], av1, bv1);
    }
  }

#pragma unroll
  for (int tm = 0; tm < 2; ++tm)
#pragma unroll
    for (int tn = 0; tn < 2; ++tn)
#pragma unroll
      for (int j = 0; j < 8; ++j) {
        int m = t0 + tm * 16 + j + kh * 8;
        int n = n0 + tn * 16 + hl;
        float v = acc[tm][tn][j];
        v = v > 0.f ? v : 0.f;
        H2[(size_t)m * I_ + n] = (__bf16)v;
      }
}

// ---------------- fused cumsum / scale-shift / center-norm epilogue ----------------
__global__ void final_norm_kernel(const float* __restrict__ H3, const float* __restrict__ divisor,
                                  float* __restrict__ out) {
  int t = blockIdx.x;
  int b = t / S_, s = t % S_;
  const float* d = H3 + (size_t)t * GF_;
  int lane = threadIdx.x & 31, wv = threadIdx.x >> 5;
  int base = threadIdx.x * 4;

  float loc[4];
  float run = 0.f;
#pragma unroll
  for (int i = 0; i < 4; ++i) { run += d[base + i]; loc[i] = run; }
  float tot = run;
  float sc = tot;
#pragma unroll
  for (int o2 = 1; o2 < 32; o2 <<= 1) {             // wave32 inclusive scan
    float nb = __shfl_up(sc, o2, 32);
    if (lane >= o2) sc += nb;
  }
  __shared__ float waveTot[8], waveOff[8];
  if (lane == 31) waveTot[wv] = sc;
  __syncthreads();
  if (threadIdx.x == 0) {
    float r = 0.f;
    for (int i = 0; i < 8; ++i) { waveOff[i] = r; r += waveTot[i]; }
  }
  __syncthreads();
  float prefix = waveOff[wv] + (sc - tot);

  float inv_div = 1.f / divisor[s];
  float vals[4], s1 = 0.f, s2 = 0.f;
#pragma unroll
  for (int i = 0; i < 4; ++i) {
    int f = base + i;
    float v = (prefix + loc[i]) * inv_div * d[F_ + f] + d[2 * F_ + f];
    vals[i] = v; s1 += v; s2 += v * v;
  }
#pragma unroll
  for (int o2 = 16; o2; o2 >>= 1) { s1 += __shfl_xor(s1, o2, 32); s2 += __shfl_xor(s2, o2, 32); }
  __shared__ float r1s[8], r2s[8];
  __shared__ float sMean, sScale;
  if (lane == 0) { r1s[wv] = s1; r2s[wv] = s2; }
  __syncthreads();
  if (threadIdx.x == 0) {
    float a = 0.f, q = 0.f;
    for (int i = 0; i < 8; ++i) { a += r1s[i]; q += r2s[i]; }
    float mean = a / (float)F_;
    float ssq = q - (float)F_ * mean * mean;
    ssq = ssq > 0.f ? ssq : 0.f;
    float denom = sqrtf(ssq) * 0.03125f + 1e-5f;    // * F^-0.5 (F=1024)
    sMean = mean;
    sScale = 0.25f / denom;                          // fold INIT_SCALE
  }
  __syncthreads();
#pragma unroll
  for (int i = 0; i < 4; ++i) {
    int f = base + i;
    out[((size_t)b * F_ + f) * S_ + s] = (vals[i] - sMean) * sScale;
  }
}

// ---------------- launch ----------------

extern "C" void kernel_launch(void* const* d_in, const int* in_sizes, int n_in,
                              void* d_out, int out_size, void* d_ws, size_t ws_size,
                              hipStream_t stream) {
  const float* inp     = (const float*)d_in[0];   // [B,F,S]
  const float* divisor = (const float*)d_in[1];   // [S]
  const float* w0_gate = (const float*)d_in[2];   // [E,F]
  const float* w0      = (const float*)d_in[3];   // [E,F,I]
  const float* w1      = (const float*)d_in[4];   // [I,I,K]
  const float* w2_gate = (const float*)d_in[5];   // [E,I]
  const float* w2      = (const float*)d_in[6];   // [E,I,3F]
  float* out = (float*)d_out;

  char* ws = (char*)d_ws;
  size_t off = 0;
  auto take = [&](size_t bytes) -> void* {
    void* p = ws + off;
    off += (bytes + 255) & ~(size_t)255;
    return p;
  };
  int*    e0_idx = (int*)take(sizeof(int) * T_);
  int*    e2_idx = (int*)take(sizeof(int) * T_);
  __bf16* xbf    = (__bf16*)take((size_t)2 * T_ * F_);
  __bf16* w0t    = (__bf16*)take((size_t)2 * E_ * F_ * I_);
  __bf16* h1bf   = (__bf16*)take((size_t)2 * T_ * I_);
  __bf16* w1t    = (__bf16*)take((size_t)2 * K_ * I_ * I_);
  __bf16* h2bf   = (__bf16*)take((size_t)2 * T_ * I_);
  __bf16* w2t    = (__bf16*)take((size_t)2 * E_ * I_ * GF_);
  float*  h3     = (float*)take(sizeof(float) * (size_t)T_ * GF_);
  (void)ws_size; (void)in_sizes; (void)n_in; (void)out_size;

  dim3 blk(256);
  cvt_x_tr_kernel<<<(unsigned)(((size_t)T_ * F_) / 256), blk, 0, stream>>>(inp, xbf);
  cvt_w_tr_kernel<<<(unsigned)(((size_t)E_ * F_ * I_) / 256), blk, 0, stream>>>(w0, w0t, F_, I_);
  cvt_w1_tr_kernel<<<(unsigned)(((size_t)K_ * I_ * I_) / 256), blk, 0, stream>>>(w1, w1t);
  cvt_w_tr_kernel<<<(unsigned)(((size_t)E_ * I_ * GF_) / 256), blk, 0, stream>>>(w2, w2t, I_, GF_);

  argmax0_kernel<<<T_ / 8, blk, 0, stream>>>(inp, w0_gate, e0_idx);
  moe_gemm_kernel<<<dim3(T_ / 32, I_ / 256), blk, 0, stream>>>(xbf, w0t, e0_idx,
                                                               h1bf, nullptr, F_, I_);
  conv_gemm_kernel<<<dim3(T_ / 32, I_ / 256), blk, 0, stream>>>(h1bf, w1t, h2bf);
  argmax2_kernel<<<T_ / 8, blk, 0, stream>>>(h2bf, w2_gate, e2_idx);
  moe_gemm_kernel<<<dim3(T_ / 32, GF_ / 256), blk, 0, stream>>>(h2bf, w2t, e2_idx,
                                                                nullptr, h3, I_, GF_);
  final_norm_kernel<<<T_, blk, 0, stream>>>(h3, divisor, out);
}
